// SuperMPNN_47974784696387
// MI455X (gfx1250) — matrix operand order repaired and checked by reading, not compiled, not run on hardware
//
#include <hip/hip_runtime.h>

// ---------------- problem constants (match reference) ----------------
#define N_   10000
#define E_   20000
#define F_   92
#define FE_  41
#define H_   80
#define HP_  96          // H padded for K-tiling (3 blocks of 32 per j)
#define KTOT (H_ * HP_)  // 7680 flattened K per i-column
#define G_   64
#define L_   3
#define D_   2
#define GROUPS_ 10
#define GSZ_  (H_ / GROUPS_)   // 8
#define EPS_  1e-5f

typedef __attribute__((ext_vector_type(16))) __bf16 v16bf;
typedef __attribute__((ext_vector_type(8)))  float  v8f;

// ---------------- small utility kernels ----------------
__global__ void k_zero(float* p, long n) {
  long t = (long)blockIdx.x * blockDim.x + threadIdx.x;
  if (t < n) p[t] = 0.f;
}

// Y[r,h] = sum_f X[r,f] * W[f,h] + b[h]   (W row-major [K][80])
__global__ void k_gemm_bias(const float* __restrict__ X, const float* __restrict__ W,
                            const float* __restrict__ b, float* __restrict__ Y,
                            int rows, int K) {
  long t = (long)blockIdx.x * blockDim.x + threadIdx.x;
  if (t >= (long)rows * H_) return;
  int r = (int)(t / H_), h = (int)(t % H_);
  const float* xr = X + (long)r * K;
  float acc = b[h];
  for (int f = 0; f < K; ++f) acc = fmaf(xr[f], W[f * H_ + h], acc);
  Y[t] = acc;
}

// GroupNorm (10 groups of 8, biased var) + ReLU
__global__ void k_gn_relu(const float* __restrict__ Y, const float* __restrict__ gamma,
                          const float* __restrict__ beta, float* __restrict__ O, int rows) {
  long t = (long)blockIdx.x * blockDim.x + threadIdx.x;
  if (t >= (long)rows * GROUPS_) return;
  int r = (int)(t / GROUPS_), g = (int)(t % GROUPS_);
  const float* p = Y + (long)r * H_ + g * GSZ_;
  float m = 0.f;
  #pragma unroll
  for (int c = 0; c < GSZ_; ++c) m += p[c];
  m *= (1.f / GSZ_);
  float v = 0.f;
  #pragma unroll
  for (int c = 0; c < GSZ_; ++c) { float d = p[c] - m; v = fmaf(d, d, v); }
  v *= (1.f / GSZ_);
  float inv = rsqrtf(v + EPS_);
  #pragma unroll
  for (int c = 0; c < GSZ_; ++c) {
    int h = g * GSZ_ + c;
    float xn = (p[c] - m) * inv;
    O[(long)r * H_ + h] = fmaxf(fmaf(xn, gamma[h], beta[h]), 0.f);
  }
}

// t[e,k] = relu(edge_h[e,:]@W1l[:,k] + b1l[k]) for k<80; t[e,80]=1; t[e,81..95]=0
__global__ void k_t_relu(const float* __restrict__ edge_h, const float* __restrict__ W1l,
                         const float* __restrict__ b1l, float* __restrict__ tbuf) {
  long t = (long)blockIdx.x * blockDim.x + threadIdx.x;
  if (t >= (long)E_ * HP_) return;
  int e = (int)(t / HP_), k = (int)(t % HP_);
  float v;
  if (k < H_) {
    const float* row = edge_h + (long)e * H_;
    float acc = b1l[k];
    for (int f = 0; f < H_; ++f) acc = fmaf(row[f], W1l[f * H_ + k], acc);
    v = fmaxf(acc, 0.f);
  } else {
    v = (k == H_) ? 1.0f : 0.0f;  // bias slot / zero pad
  }
  tbuf[t] = v;
}

// Pack W2[l] (+ b2[l] into padded k==80 slot) into bf16, transposed:
// w2bt[i][j*96 + k] = bf16(W2l[k, i*80 + j]); k==80 -> b2l[i*80+j]; k>80 -> 0
__global__ void k_w2_pack(const float* __restrict__ W2l, const float* __restrict__ b2l,
                          unsigned short* __restrict__ w2bt) {
  long t = (long)blockIdx.x * blockDim.x + threadIdx.x;
  if (t >= (long)H_ * KTOT) return;
  int i = (int)(t / KTOT);
  int rem = (int)(t % KTOT);
  int j = rem / HP_, k = rem % HP_;
  float v = 0.f;
  if (k < H_)       v = W2l[(long)k * (H_ * H_) + i * H_ + j];
  else if (k == H_) v = b2l[i * H_ + j];
  __bf16 h = (__bf16)v;
  w2bt[t] = __builtin_bit_cast(unsigned short, h);
}

// ---------------- fused edge-message WMMA kernel ----------------
// msgs[e,i] = sum_{j,k} x[e,j]*t[e,k]*W2[k,i*80+j]  (+ bias folded via k==80)
// One block = 32 edges (2 tiles of 16); 5 waves = 5 tiles of 16 output channels.
// Each wave processes BOTH edge tiles so every B fragment is used twice, and
// B rows are double-buffered (prefetched one j-iteration ahead).
__global__ __launch_bounds__(160) void
k_msgs_wmma(const float* __restrict__ tbuf, const float* __restrict__ node_h,
            const int* __restrict__ src, const unsigned short* __restrict__ w2bt,
            float* __restrict__ msgs) {
  __shared__ float t_s[32 * 97];   // padded stride vs 64 banks
  __shared__ float x_s[32 * 81];

  const int e0 = blockIdx.x * 32;
  for (int idx = threadIdx.x; idx < 32 * HP_; idx += blockDim.x) {
    int m = idx / HP_, k = idx % HP_;
    t_s[m * 97 + k] = tbuf[(long)(e0 + m) * HP_ + k];
  }
  for (int idx = threadIdx.x; idx < 32 * H_; idx += blockDim.x) {
    int m = idx / H_, j = idx % H_;
    int s = src[e0 + m];
    x_s[m * 81 + j] = node_h[(long)s * H_ + j];
  }
  __syncthreads();

  const int tid  = threadIdx.x;
  const int wave = tid >> 5;        // 0..4 -> output-channel tile
  const int lane = tid & 31;
  const int m    = lane & 15;       // edge row within tile (A), column sel (B/C/D)
  const int hf   = lane >> 4;       // lane half selects K sub-ranges / M+8 rows
  const int icol = wave * 16 + m;   // output channel for this lane (B/C/D column)

  // Per-lane t fragments, pre-converted to bf16 (constant over the j loop).
  // A-layout: lanes 0-15 hold K {0..15}, lanes 16-31 hold {16..31}.
  v16bf tf[2][3];
  #pragma unroll
  for (int tile = 0; tile < 2; ++tile) {
    const float* trow = t_s + (tile * 16 + m) * 97;
    #pragma unroll
    for (int kb = 0; kb < 3; ++kb) {
      const int cb = kb * 32 + hf * 16;
      #pragma unroll
      for (int q = 0; q < 8; ++q) {
        tf[tile][kb][q]     = (__bf16)trow[cb + q];
        tf[tile][kb][q + 8] = (__bf16)trow[cb + 8 + q];
      }
    }
  }

  const unsigned short* bptr = w2bt + (long)icol * KTOT + hf * 16;
  // preload j=0 B fragments (32x16 bf16 each; lane half picks 16 consecutive K)
  v16bf bc0 = *reinterpret_cast<const v16bf*>(bptr);
  v16bf bc1 = *reinterpret_cast<const v16bf*>(bptr + 32);
  v16bf bc2 = *reinterpret_cast<const v16bf*>(bptr + 64);

  v8f acc0 = {}, acc1 = {};
  for (int j = 0; j < H_; ++j) {
    // prefetch next j's B fragments (w2bt over-allocated by one row -> safe at j==79)
    const unsigned short* bp2 = bptr + HP_;
    v16bf bn0 = *reinterpret_cast<const v16bf*>(bp2);
    v16bf bn1 = *reinterpret_cast<const v16bf*>(bp2 + 32);
    v16bf bn2 = *reinterpret_cast<const v16bf*>(bp2 + 64);

    const __bf16 xb0 = (__bf16)x_s[m * 81 + j];
    const __bf16 xb1 = (__bf16)x_s[(16 + m) * 81 + j];
    v16bf xv0, xv1;
    #pragma unroll
    for (int q = 0; q < 16; ++q) { xv0[q] = xb0; xv1[q] = xb1; }

    // A = t (bf16) * x-broadcast (bf16) -> v_pk_mul_bf16; f32 accumulate in WMMA.
    // Alternate acc0/acc1 so consecutive WMMAs have no D->A/B dependency.
    acc0 = __builtin_amdgcn_wmma_f32_16x16x32_bf16(false, tf[0][0] * xv0, false, bc0,
                                                   (short)0, acc0, false, false);
    acc1 = __builtin_amdgcn_wmma_f32_16x16x32_bf16(false, tf[1][0] * xv1, false, bc0,
                                                   (short)0, acc1, false, false);
    acc0 = __builtin_amdgcn_wmma_f32_16x16x32_bf16(false, tf[0][1] * xv0, false, bc1,
                                                   (short)0, acc0, false, false);
    acc1 = __builtin_amdgcn_wmma_f32_16x16x32_bf16(false, tf[1][1] * xv1, false, bc1,
                                                   (short)0, acc1, false, false);
    acc0 = __builtin_amdgcn_wmma_f32_16x16x32_bf16(false, tf[0][2] * xv0, false, bc2,
                                                   (short)0, acc0, false, false);
    acc1 = __builtin_amdgcn_wmma_f32_16x16x32_bf16(false, tf[1][2] * xv1, false, bc2,
                                                   (short)0, acc1, false, false);

    bc0 = bn0; bc1 = bn1; bc2 = bn2;
    bptr += HP_;
  }

  // D layout: VGPR r -> row M = r (+8 for upper lane half), column = icol
  #pragma unroll
  for (int r = 0; r < 8; ++r) {
    msgs[(long)(e0 + r + hf * 8) * H_ + icol]      = acc0[r];
    msgs[(long)(e0 + 16 + r + hf * 8) * H_ + icol] = acc1[r];
  }
}

// ---------------- segment ops ----------------
__global__ void k_count(const int* __restrict__ idx, float* __restrict__ cnt, int n) {
  int t = blockIdx.x * blockDim.x + threadIdx.x;
  if (t < n) atomicAdd(&cnt[idx[t]], 1.0f);
}

__global__ void k_scatter_rows(const float* __restrict__ in, const int* __restrict__ idx,
                               float* __restrict__ out, int rows) {
  long t = (long)blockIdx.x * blockDim.x + threadIdx.x;
  if (t >= (long)rows * H_) return;
  int r = (int)(t / H_), i = (int)(t % H_);
  atomicAdd(&out[(long)idx[r] * H_ + i], in[t]);
}

__global__ void k_scale_rows(float* __restrict__ feat, const float* __restrict__ cnt, int rows) {
  long t = (long)blockIdx.x * blockDim.x + threadIdx.x;
  if (t >= (long)rows * H_) return;
  int r = (int)(t / H_);
  feat[t] *= 1.0f / fmaxf(cnt[r], 1.0f);
}

// ---------------- GRU update (keras reset_after, gates z,r,h) ----------------
__global__ void k_gru(const float* __restrict__ agg, const float* __restrict__ cnt,
                      const float* __restrict__ h_in, float* __restrict__ h_out,
                      float* __restrict__ node_h,
                      const float* __restrict__ Wk, const float* __restrict__ Wr,
                      const float* __restrict__ b_in, const float* __restrict__ b_rec) {
  long t = (long)blockIdx.x * blockDim.x + threadIdx.x;
  if (t >= (long)N_ * H_) return;
  int n = (int)(t / H_), i = (int)(t % H_);
  const float scale = 1.0f / fmaxf(cnt[n], 1.0f);
  const float* a = agg  + (long)n * H_;
  const float* h = h_in + (long)n * H_;
  float xz = 0.f, xr = 0.f, xh = 0.f, hz = 0.f, hr = 0.f, hh = 0.f;
  for (int k = 0; k < H_; ++k) {
    const float av = a[k], hv = h[k];
    const float* wk = Wk + k * (3 * H_);
    const float* wr = Wr + k * (3 * H_);
    xz = fmaf(av, wk[i], xz);         xr = fmaf(av, wk[H_ + i], xr);
    xh = fmaf(av, wk[2 * H_ + i], xh);
    hz = fmaf(hv, wr[i], hz);         hr = fmaf(hv, wr[H_ + i], hr);
    hh = fmaf(hv, wr[2 * H_ + i], hh);
  }
  xz = fmaf(xz, scale, b_in[i]);
  xr = fmaf(xr, scale, b_in[H_ + i]);
  xh = fmaf(xh, scale, b_in[2 * H_ + i]);
  hz += b_rec[i]; hr += b_rec[H_ + i]; hh += b_rec[2 * H_ + i];
  const float z  = 1.f / (1.f + expf(-(xz + hz)));
  const float r  = 1.f / (1.f + expf(-(xr + hr)));
  const float hc = tanhf(fmaf(r, hh, xh));
  const float hv = h[i];
  const float hnew = fmaf(z, hv - hc, hc);  // z*h + (1-z)*hc
  h_out[t]  = hnew;
  node_h[t] = hnew + node_h[t];             // node_h becomes new residual
}

// ---------------- head output ----------------
__global__ void k_head_out(const float* __restrict__ hid, const float* __restrict__ Wo,
                           const float* __restrict__ bo, float* __restrict__ out) {
  int g = blockIdx.x * blockDim.x + threadIdx.x;
  if (g >= G_) return;
  float acc = bo[0];
  for (int i = 0; i < H_; ++i) acc = fmaf(hid[(long)g * H_ + i], Wo[i], acc);
  out[g] = acc;
}

// ---------------- host orchestration ----------------
static inline dim3 gs(long n) { return dim3((unsigned)((n + 255) / 256)); }

extern "C" void kernel_launch(void* const* d_in, const int* in_sizes, int n_in,
                              void* d_out, int out_size, void* d_ws, size_t ws_size,
                              hipStream_t stream) {
  (void)in_sizes; (void)n_in; (void)out_size; (void)ws_size;
  const float* node_features = (const float*)d_in[0];
  const float* edge_features = (const float*)d_in[1];
  const int*   edge_indices  = (const int*)d_in[2];
  const int*   graph_indices = (const int*)d_in[3];
  const float* W_ne = (const float*)d_in[4];  const float* b_ne = (const float*)d_in[5];
  const float* gamma_n = (const float*)d_in[6]; const float* beta_n = (const float*)d_in[7];
  const float* W_ee = (const float*)d_in[8];  const float* b_ee = (const float*)d_in[9];
  const float* gamma_e = (const float*)d_in[10]; const float* beta_e = (const float*)d_in[11];
  const float* W1 = (const float*)d_in[12];   const float* b1 = (const float*)d_in[13];
  const float* W2 = (const float*)d_in[14];   const float* b2 = (const float*)d_in[15];
  const float* Wk = (const float*)d_in[16];   const float* Wr = (const float*)d_in[17];
  const float* b_in = (const float*)d_in[18]; const float* b_rec = (const float*)d_in[19];
  const float* Wd = (const float*)d_in[20];   const float* bd = (const float*)d_in[21];
  const float* gamma_d = (const float*)d_in[22]; const float* beta_d = (const float*)d_in[23];
  const float* Wo = (const float*)d_in[24];   const float* bo = (const float*)d_in[25];
  const int* src = edge_indices;       // edge_indices[0]
  const int* dst = edge_indices + E_;  // edge_indices[1]
  float* out = (float*)d_out;

  // carve workspace (256B aligned)
  char* base = (char*)d_ws;
  size_t off = 0;
  auto alloc = [&](size_t bytes) -> void* {
    void* p = base + off;
    off = (off + bytes + 255) & ~(size_t)255;
    return p;
  };
  float* node_h = (float*)alloc((size_t)N_ * H_ * 4);
  float* h_a    = (float*)alloc((size_t)N_ * H_ * 4);
  float* h_b    = (float*)alloc((size_t)N_ * H_ * 4);
  float* edge_h = (float*)alloc((size_t)E_ * H_ * 4);
  float* tbuf   = (float*)alloc((size_t)E_ * HP_ * 4);
  float* msgs   = (float*)alloc((size_t)E_ * H_ * 4);
  float* agg    = (float*)alloc((size_t)N_ * H_ * 4);
  float* cnt    = (float*)alloc((size_t)N_ * 4);
  // +HP_ over-allocation: the j==79 B prefetch reads one row past the end
  unsigned short* w2bt = (unsigned short*)alloc(((size_t)H_ * KTOT + HP_) * 2);
  float* ybuf   = (float*)alloc((size_t)E_ * H_ * 4);  // pre-norm scratch (>= N*H)
  float* gfeat  = (float*)alloc((size_t)G_ * H_ * 4);
  float* gcnt   = (float*)alloc((size_t)G_ * 4);
  float* ha     = (float*)alloc((size_t)G_ * H_ * 4);
  float* hb     = (float*)alloc((size_t)G_ * H_ * 4);

  // embeddings: Dense + GroupNorm + ReLU
  k_gemm_bias<<<gs((long)N_ * H_), 256, 0, stream>>>(node_features, W_ne, b_ne, ybuf, N_, F_);
  k_gn_relu<<<gs((long)N_ * GROUPS_), 256, 0, stream>>>(ybuf, gamma_n, beta_n, node_h, N_);
  k_gemm_bias<<<gs((long)E_ * H_), 256, 0, stream>>>(edge_features, W_ee, b_ee, ybuf, E_, FE_);
  k_gn_relu<<<gs((long)E_ * GROUPS_), 256, 0, stream>>>(ybuf, gamma_e, beta_e, edge_h, E_);

  // init GRU state + in-degree counts (dst is layer-invariant)
  k_zero<<<gs((long)N_ * H_), 256, 0, stream>>>(h_a, (long)N_ * H_);
  k_zero<<<gs(N_), 256, 0, stream>>>(cnt, N_);
  k_count<<<gs(E_), 256, 0, stream>>>(dst, cnt, E_);

  float* h_cur = h_a;
  float* h_nxt = h_b;
  for (int l = 0; l < L_; ++l) {
    const float* W1l = W1 + (size_t)l * H_ * H_;
    const float* b1l = b1 + (size_t)l * H_;
    const float* W2l = W2 + (size_t)l * H_ * H_ * H_;
    const float* b2l = b2 + (size_t)l * H_ * H_;
    const float* Wkl = Wk + (size_t)l * H_ * 3 * H_;
    const float* Wrl = Wr + (size_t)l * H_ * 3 * H_;
    const float* bil = b_in + (size_t)l * 3 * H_;
    const float* brl = b_rec + (size_t)l * 3 * H_;

    k_t_relu<<<gs((long)E_ * HP_), 256, 0, stream>>>(edge_h, W1l, b1l, tbuf);
    k_w2_pack<<<gs((long)H_ * KTOT), 256, 0, stream>>>(W2l, b2l, w2bt);
    k_msgs_wmma<<<dim3(E_ / 32), dim3(160), 0, stream>>>(tbuf, node_h, src, w2bt, msgs);

    k_zero<<<gs((long)N_ * H_), 256, 0, stream>>>(agg, (long)N_ * H_);
    k_scatter_rows<<<gs((long)E_ * H_), 256, 0, stream>>>(msgs, dst, agg, E_);

    k_gru<<<gs((long)N_ * H_), 256, 0, stream>>>(agg, cnt, h_cur, h_nxt, node_h,
                                                 Wkl, Wrl, bil, brl);
    float* tmp = h_cur; h_cur = h_nxt; h_nxt = tmp;
  }

  // mean-pool nodes -> graphs
  k_zero<<<gs((long)G_ * H_), 256, 0, stream>>>(gfeat, (long)G_ * H_);
  k_zero<<<gs(G_), 256, 0, stream>>>(gcnt, G_);
  k_count<<<gs(N_), 256, 0, stream>>>(graph_indices, gcnt, N_);
  k_scatter_rows<<<gs((long)N_ * H_), 256, 0, stream>>>(node_h, graph_indices, gfeat, N_);
  k_scale_rows<<<gs((long)G_ * H_), 256, 0, stream>>>(gfeat, gcnt, G_);

  // dense head
  const float* hin = gfeat;
  for (int d = 0; d < D_; ++d) {
    k_gemm_bias<<<gs((long)G_ * H_), 256, 0, stream>>>(hin, Wd + (size_t)d * H_ * H_,
                                                       bd + (size_t)d * H_, ha, G_, H_);
    k_gn_relu<<<gs((long)G_ * GROUPS_), 256, 0, stream>>>(ha, gamma_d + (size_t)d * H_,
                                                          beta_d + (size_t)d * H_, hb, G_);
    hin = hb;
  }
  k_head_out<<<dim3(1), dim3(64), 0, stream>>>(hin, Wo, bo, out);
}
